// TubeProposalNetwork_64622077936311
// MI455X (gfx1250) — matrix operand order repaired and checked by reading, not compiled, not run on hardware
//
#include <hip/hip_runtime.h>
#include <hip/hip_bf16.h>
#include <math.h>

// ---------------- problem constants ----------------
#define N_ROI   1024
#define C_FEAT  256
#define H_FEAT  32
#define W_FEAT  32
#define OUTSZ   7
#define D_HALF  (C_FEAT * OUTSZ * OUTSZ)   // 12544
#define D_IN    (2 * D_HALF)               // 25088
#define D_HID   512
#define D_OUT   11
#define K_HALF  (D_IN / 2)                 // 12544 (divisible by 64)
#define ROI_SCALE 0.03125f                 // 1/32

typedef __attribute__((ext_vector_type(16))) __bf16 v16bf;
typedef __attribute__((ext_vector_type(8)))  float  v8f;

union Frag16 {            // 16 bf16 = 32 bytes = 8 VGPRs; filled via two 16B loads
    v16bf v;
    uint4 q[2];
};

__device__ __forceinline__ unsigned short f2bf_rne(float x) {
    unsigned int u = __float_as_uint(x);
    unsigned int r = (u + 0x7FFFu + ((u >> 16) & 1u)) >> 16;
    return (unsigned short)r;
}

// ---------------------------------------------------------------------------
// K1: RoIAlign (torchvision-style, aligned=false) fused over both feature maps,
//     emitting the bf16 activation matrix A[N_ROI][D_IN] (row-major, k-major).
// ---------------------------------------------------------------------------
__global__ void roi_align_bf16_kernel(const float* __restrict__ feats1,
                                      const float* __restrict__ feats2,
                                      const float* __restrict__ props,
                                      unsigned short* __restrict__ A) {
    int idx = blockIdx.x * blockDim.x + threadIdx.x;
    if (idx >= N_ROI * D_IN) return;
    int n = idx / D_IN;
    int j = idx - n * D_IN;

    const float* feat = (j < D_HALF) ? feats1 : feats2;
    int jj = (j < D_HALF) ? j : (j - D_HALF);
    int c  = jj / (OUTSZ * OUTSZ);
    int s  = jj - c * (OUTSZ * OUTSZ);
    int py = s / OUTSZ;
    int px = s - py * OUTSZ;

    float x1 = props[n * 4 + 0] * ROI_SCALE;
    float y1 = props[n * 4 + 1] * ROI_SCALE;
    float x2 = props[n * 4 + 2] * ROI_SCALE;
    float y2 = props[n * 4 + 3] * ROI_SCALE;
    float rw = fmaxf(x2 - x1, 1.0f);
    float rh = fmaxf(y2 - y1, 1.0f);
    float bw = rw * (1.0f / OUTSZ);
    float bh = rh * (1.0f / OUTSZ);

    const float* fc = feat + c * (H_FEAT * W_FEAT);

    float sum = 0.0f;
    #pragma unroll
    for (int sy = 0; sy < 2; ++sy) {
        float oy = y1 + (float)py * bh + ((float)sy + 0.5f) * (bh * 0.5f);
        bool  vy = (oy >= -1.0f) && (oy <= (float)H_FEAT);
        float oyc = fminf(fmaxf(oy, 0.0f), (float)(H_FEAT - 1));
        int   y0  = (int)floorf(oyc);
        int   y1i = min(y0 + 1, H_FEAT - 1);
        float ly  = oyc - (float)y0;
        #pragma unroll
        for (int sx = 0; sx < 2; ++sx) {
            float ox = x1 + (float)px * bw + ((float)sx + 0.5f) * (bw * 0.5f);
            bool  vx = (ox >= -1.0f) && (ox <= (float)W_FEAT);
            float oxc = fminf(fmaxf(ox, 0.0f), (float)(W_FEAT - 1));
            int   x0  = (int)floorf(oxc);
            int   x1i = min(x0 + 1, W_FEAT - 1);
            float lx  = oxc - (float)x0;
            float v = fc[y0  * W_FEAT + x0 ] * (1.0f - ly) * (1.0f - lx)
                    + fc[y0  * W_FEAT + x1i] * (1.0f - ly) * lx
                    + fc[y1i * W_FEAT + x0 ] * ly * (1.0f - lx)
                    + fc[y1i * W_FEAT + x1i] * ly * lx;
            sum += (vy && vx) ? v : 0.0f;
        }
    }
    A[idx] = f2bf_rne(sum * 0.25f);
}

// ---------------------------------------------------------------------------
// K2: W1 [D_IN, D_HID] f32 -> Wt [D_HID, D_IN] bf16 via LDS-tiled transpose.
//     32x32 tiles, +1 pad column to avoid LDS bank conflicts; both the global
//     read and the global write are coalesced.
// ---------------------------------------------------------------------------
__global__ void w1_transpose_bf16_kernel(const float* __restrict__ W1,
                                         unsigned short* __restrict__ Wt) {
    __shared__ unsigned short tile[32][33];
    int kb = blockIdx.x * 32;          // k-tile base (784 tiles)
    int nb = blockIdx.y * 32;          // n-tile base (16 tiles)
    int tx = threadIdx.x;              // 0..31
    int ty = threadIdx.y;              // 0..7

    #pragma unroll
    for (int i = ty; i < 32; i += 8) {
        // coalesced read of W1 row (kb+i), cols nb+tx; store transposed in LDS
        tile[tx][i] = f2bf_rne(W1[(size_t)(kb + i) * D_HID + nb + tx]);
    }
    __syncthreads();
    #pragma unroll
    for (int i = ty; i < 32; i += 8) {
        // coalesced write of Wt row (nb+i), cols kb+tx
        Wt[(size_t)(nb + i) * D_IN + kb + tx] = tile[i][tx];
    }
}

// ---------------------------------------------------------------------------
// K3: partial GEMM  P[s] = A[:, ks:ks+K_HALF] @ W1bf[ks:ks+K_HALF, :]
//     via v_wmma_f32_16x16x32_bf16, 2x2 register blocking (32x32 C per wave)
//     to halve L2 operand traffic (1 KB loaded per WMMA), K split in two for
//     2x wave parallelism (1024 waves).
//     Fragment layouts per CDNA5 ISA 7.12.2:
//       A (16x32 bf16): lane m=lane&15; hi=lane>>4 selects K {+8,+24} halves.
//       B (32x16 bf16): lane n=lane&15; hi=lane>>4 selects K {+8,+24} halves.
//       C/D (16x16 f32): VGPR r -> row r + 8*hi, col lane&15.
// ---------------------------------------------------------------------------
__global__ void gemm1_wmma_kernel(const unsigned short* __restrict__ A,
                                  const unsigned short* __restrict__ Wt,
                                  float* __restrict__ P) {   // P: [2][1024][512]
    int wave = (blockIdx.x * blockDim.x + threadIdx.x) >> 5;  // 0..1023
    int lane = threadIdx.x & 31;
    int s    = wave >> 9;              // K-split half: 0/1
    int rem  = wave & 511;
    int mg   = rem >> 4;               // 0..31  (32-row blocks)
    int ng   = rem & 15;               // 0..15  (32-col blocks)
    int m    = lane & 15;
    int hi   = lane >> 4;              // 0 or 1

    const unsigned short* A0 = A  + (size_t)(mg * 32 + m) * D_IN + hi * 8  + s * K_HALF;
    const unsigned short* A1 = A0 + (size_t)16 * D_IN;
    const unsigned short* B0 = Wt + (size_t)(ng * 32 + m) * D_IN + hi * 8  + s * K_HALF;
    const unsigned short* B1 = B0 + (size_t)16 * D_IN;

    v8f acc00 = {}, acc01 = {}, acc10 = {}, acc11 = {};
    #pragma unroll 2
    for (int k = 0; k < K_HALF; k += 32) {
        Frag16 a0, a1, b0, b1;
        a0.q[0] = *(const uint4*)(A0 + k);
        a0.q[1] = *(const uint4*)(A0 + k + 16);
        a1.q[0] = *(const uint4*)(A1 + k);
        a1.q[1] = *(const uint4*)(A1 + k + 16);
        b0.q[0] = *(const uint4*)(B0 + k);
        b0.q[1] = *(const uint4*)(B0 + k + 16);
        b1.q[0] = *(const uint4*)(B1 + k);
        b1.q[1] = *(const uint4*)(B1 + k + 16);
        acc00 = __builtin_amdgcn_wmma_f32_16x16x32_bf16(false, a0.v, false, b0.v, (short)0, acc00, false, false);
        acc01 = __builtin_amdgcn_wmma_f32_16x16x32_bf16(false, a0.v, false, b1.v, (short)0, acc01, false, false);
        acc10 = __builtin_amdgcn_wmma_f32_16x16x32_bf16(false, a1.v, false, b0.v, (short)0, acc10, false, false);
        acc11 = __builtin_amdgcn_wmma_f32_16x16x32_bf16(false, a1.v, false, b1.v, (short)0, acc11, false, false);
    }

    float* Ps = P + (size_t)s * N_ROI * D_HID;
    int col0 = ng * 32 + (lane & 15);
    #pragma unroll
    for (int r = 0; r < 8; ++r) {
        int row0 = mg * 32 + r + 8 * hi;
        Ps[(size_t)row0        * D_HID + col0     ] = acc00[r];
        Ps[(size_t)row0        * D_HID + col0 + 16] = acc01[r];
        Ps[(size_t)(row0 + 16) * D_HID + col0     ] = acc10[r];
        Ps[(size_t)(row0 + 16) * D_HID + col0 + 16] = acc11[r];
    }
}

// ---------------------------------------------------------------------------
// K3b: H = relu(P0 + P1 + b1)   (combine K-split partials, fuse bias + ReLU)
// ---------------------------------------------------------------------------
__global__ void reduce_bias_relu_kernel(const float* __restrict__ P,
                                        const float* __restrict__ b1,
                                        float* __restrict__ H) {
    int idx = blockIdx.x * blockDim.x + threadIdx.x;
    if (idx >= N_ROI * D_HID) return;
    float v = P[idx] + P[N_ROI * D_HID + idx] + b1[idx & (D_HID - 1)];
    H[idx] = v > 0.0f ? v : 0.0f;
}

// ---------------------------------------------------------------------------
// K4: out = H @ W2 + b2, sigmoid/scatter into concatenated output:
//     [0,1024)      obj0 = sigmoid(out[:,0])
//     [1024,2048)   obj1 = sigmoid(out[:,1])
//     [2048,6144)   boxes0 = out[:,2:6]
//     [6144,10240)  boxes1 = out[:,6:10]
//     [10240,11264) id = sigmoid(out[:,10])
// ---------------------------------------------------------------------------
__global__ void fc2_out_kernel(const float* __restrict__ H,
                               const float* __restrict__ W2,
                               const float* __restrict__ b2,
                               float* __restrict__ out) {
    int n = blockIdx.x;        // 0..1023
    int mcol = threadIdx.x;    // 0..31
    if (mcol >= D_OUT) return;
    const float* h = H + (size_t)n * D_HID;
    float acc = b2[mcol];
    #pragma unroll 8
    for (int k = 0; k < D_HID; ++k) acc = fmaf(h[k], W2[k * D_OUT + mcol], acc);

    if (mcol == 0)      out[n]                       = 1.0f / (1.0f + expf(-acc));
    else if (mcol == 1) out[1024 + n]                = 1.0f / (1.0f + expf(-acc));
    else if (mcol <= 5) out[2048 + n * 4 + (mcol-2)] = acc;
    else if (mcol <= 9) out[6144 + n * 4 + (mcol-6)] = acc;
    else                out[10240 + n]               = 1.0f / (1.0f + expf(-acc));
}

extern "C" void kernel_launch(void* const* d_in, const int* in_sizes, int n_in,
                              void* d_out, int out_size, void* d_ws, size_t ws_size,
                              hipStream_t stream) {
    const float* feats1 = (const float*)d_in[0];
    const float* feats2 = (const float*)d_in[1];
    const float* props  = (const float*)d_in[2];
    const float* W1     = (const float*)d_in[3];
    const float* b1     = (const float*)d_in[4];
    const float* W2     = (const float*)d_in[5];
    const float* b2     = (const float*)d_in[6];
    float* out = (float*)d_out;

    // workspace layout
    unsigned short* A  = (unsigned short*)d_ws;               // 1024*25088 bf16 (51.4 MB)
    unsigned short* Wt = A + (size_t)N_ROI * D_IN;            // 512*25088 bf16 (25.7 MB)
    float* Pbuf        = (float*)(Wt + (size_t)D_HID * D_IN); // 2*1024*512 f32 ( 4.2 MB)
    float* Hbuf        = Pbuf + (size_t)2 * N_ROI * D_HID;    // 1024*512 f32   ( 2.1 MB)

    // K1: RoIAlign -> bf16 activation matrix
    {
        int total = N_ROI * D_IN;
        roi_align_bf16_kernel<<<(total + 255) / 256, 256, 0, stream>>>(feats1, feats2, props, A);
    }
    // K2: W1 -> bf16, LDS-tiled transpose to B layout
    {
        dim3 grid(D_IN / 32, D_HID / 32);   // 784 x 16
        dim3 block(32, 8);
        w1_transpose_bf16_kernel<<<grid, block, 0, stream>>>(W1, Wt);
    }
    // K3: WMMA GEMM, 2x2 blocked, K-split x2 (1024 waves = 128 blocks x 256)
    gemm1_wmma_kernel<<<128, 256, 0, stream>>>(A, Wt, Pbuf);

    // K3b: combine partials + bias + ReLU
    {
        int total = N_ROI * D_HID;
        reduce_bias_relu_kernel<<<(total + 255) / 256, 256, 0, stream>>>(Pbuf, b1, Hbuf);
    }
    // K4: tiny FC2 + sigmoid + scatter
    fc2_out_kernel<<<N_ROI, 32, 0, stream>>>(Hbuf, W2, b2, out);
}
